// MultiHeadAttention_81252191306455
// MI455X (gfx1250) — matrix operand order repaired and checked by reading, not compiled, not run on hardware
//
#include <hip/hip_runtime.h>
#include <hip/hip_bf16.h>

typedef _Float16 f16;
typedef __attribute__((ext_vector_type(16))) _Float16 v16h;
typedef __attribute__((ext_vector_type(8)))  _Float16 v8h;
typedef __attribute__((ext_vector_type(4)))  _Float16 v4h;
typedef __attribute__((ext_vector_type(8)))  float    v8f;
typedef __attribute__((ext_vector_type(4)))  float    v4f;

#define S_LEN 2048
#define E_DIM 1024
#define NHEADS 16
#define HEADD 64

static __device__ __forceinline__ v16h cat8(v8h lo, v8h hi) {
  return __builtin_shufflevector(lo, hi, 0,1,2,3,4,5,6,7,8,9,10,11,12,13,14,15);
}
static __device__ __forceinline__ v8f zero8() {
  v8f z; for (int i = 0; i < 8; ++i) z[i] = 0.0f; return z;
}
static __device__ __forceinline__ v8f wmma16(v16h a, v16h b, v8f c) {
  // D = A(16x32 f16) * B(32x16 f16) + C(16x16 f32)
  return __builtin_amdgcn_wmma_f32_16x16x32_f16(false, a, false, b, (short)0, c, false, false);
}

// ---------------------------------------------------------------------------
// Prep: fp32 -> f16 elementwise (x), vectorized 4-wide
// ---------------------------------------------------------------------------
__global__ __launch_bounds__(256) void mha_cvt_f16(const float* __restrict__ in,
                                                   f16* __restrict__ out) {
  const size_t id = (size_t)blockIdx.x * 256 + threadIdx.x;
  const v4f v = reinterpret_cast<const v4f*>(in)[id];
  v4h h;
  for (int i = 0; i < 4; ++i) h[i] = (f16)v[i];
  reinterpret_cast<v4h*>(out)[id] = h;
}

// ---------------------------------------------------------------------------
// Prep: fp32 [1024][1024] -> f16 transposed [1024][1024]  (outT[n][k]=in[k][n])
// ---------------------------------------------------------------------------
__global__ __launch_bounds__(256) void mha_transpose_f16(const float* __restrict__ in,
                                                         f16* __restrict__ outT) {
  __shared__ float tile[32][33];
  const int tx = threadIdx.x, ty = threadIdx.y;
  const int bx = blockIdx.x, by = blockIdx.y;
  for (int i = 0; i < 4; ++i)
    tile[ty + i * 8][tx] = in[(size_t)(by * 32 + ty + i * 8) * E_DIM + bx * 32 + tx];
  __syncthreads();
  for (int i = 0; i < 4; ++i)
    outT[(size_t)(bx * 32 + ty + i * 8) * E_DIM + by * 32 + tx] = (f16)tile[tx][ty + i * 8];
}

// ---------------------------------------------------------------------------
// QKV projection: [4096,1024] x (Wq|Wk|Wv)^T.
// One wave = 32x64 output tile (2 M sub-tiles x 4 N tiles): each B fragment
// is reused for 2 WMMAs out of registers -> ~21 flop/operand-byte per lane.
// q,k written as [B,H,S,D]; v written transposed [B,H,D,S].
// ---------------------------------------------------------------------------
__global__ __launch_bounds__(256) void mha_qkv_gemm(const f16* __restrict__ xh,
                                                    const f16* __restrict__ WqT,
                                                    const f16* __restrict__ WkT,
                                                    const f16* __restrict__ WvT,
                                                    f16* __restrict__ qh,
                                                    f16* __restrict__ kh,
                                                    f16* __restrict__ vT) {
  const int wid  = blockIdx.x * 8 + (threadIdx.x >> 5);   // 6144 waves
  const int lane = threadIdx.x & 31;
  const int n    = lane & 15;
  const int half = lane >> 4;
  const int ntile = wid & 15;          // head index (N tile of 64)
  const int mtile = (wid >> 4) & 127;  // 128 M-tiles of 32
  const int which = wid >> 11;         // 0=q 1=k 2=v
  const int m0 = mtile * 32;
  const int n0 = ntile * 64;
  const f16* __restrict__ W = (which == 0) ? WqT : (which == 1) ? WkT : WvT;

  v8f acc[2][4];
  for (int mm = 0; mm < 2; ++mm)
    for (int t = 0; t < 4; ++t) acc[mm][t] = zero8();

  const f16* ar0 = xh + (size_t)(m0 + n) * E_DIM;
  const f16* ar1 = xh + (size_t)(m0 + 16 + n) * E_DIM;
  for (int kk = 0; kk < E_DIM; kk += 32) {
    const v16h af0 = cat8(*reinterpret_cast<const v8h*>(ar0 + kk + half * 16),
                          *reinterpret_cast<const v8h*>(ar0 + kk + half * 16 + 8));
    const v16h af1 = cat8(*reinterpret_cast<const v8h*>(ar1 + kk + half * 16),
                          *reinterpret_cast<const v8h*>(ar1 + kk + half * 16 + 8));
    for (int t = 0; t < 4; ++t) {
      const v16h bf = *reinterpret_cast<const v16h*>(
          W + (size_t)(n0 + t * 16 + n) * E_DIM + kk + half * 16);
      acc[0][t] = wmma16(af0, bf, acc[0][t]);
      acc[1][t] = wmma16(af1, bf, acc[1][t]);
    }
  }

  for (int mm = 0; mm < 2; ++mm)
    for (int t = 0; t < 4; ++t) {
      const int d = t * 16 + n;                 // dim within head
      for (int r = 0; r < 8; ++r) {
        const int row = m0 + mm * 16 + r + 8 * half;   // global token row
        const int b = row >> 11, s = row & (S_LEN - 1);
        const size_t bh = (size_t)b * NHEADS + ntile;
        const f16 val = (f16)acc[mm][t][r];
        if (which == 0)      qh[(bh * S_LEN + s) * HEADD + d] = val;
        else if (which == 1) kh[(bh * S_LEN + s) * HEADD + d] = val;
        else                 vT[(bh * HEADD + d) * S_LEN + s] = val;
      }
    }
}

// ---------------------------------------------------------------------------
// Causal flash attention. One wave = 16 query rows of one (b,h).
// 32 keys/iter: 2 score WMMAs (K=64), online softmax (shuffle reductions in
// 16-lane halves), P via LDS -> A-layout, 4 PV WMMAs (K=32).
// ---------------------------------------------------------------------------
__global__ __launch_bounds__(256) void mha_attention(const f16* __restrict__ qh,
                                                     const f16* __restrict__ kh,
                                                     const f16* __restrict__ vT,
                                                     f16* __restrict__ oh) {
  __shared__ f16 plds[8][16][32];
  const int wave = threadIdx.x >> 5;
  const int lane = threadIdx.x & 31;
  const int n    = lane & 15;
  const int half = lane >> 4;

  const int qblk = blockIdx.x & 15;   // 16 query blocks of 128 rows
  const int bh   = blockIdx.x >> 4;   // 0..31
  const int b    = bh >> 4;
  const int hd   = bh & 15;

  const f16* __restrict__ qp = qh + (size_t)bh * S_LEN * HEADD;
  const f16* __restrict__ kp = kh + (size_t)bh * S_LEN * HEADD;
  const f16* __restrict__ vp = vT + (size_t)bh * HEADD * S_LEN;

  const int m0 = qblk * 128 + wave * 16;

  v16h aq[2];
  {
    const f16* ar = qp + (size_t)(m0 + n) * HEADD;
    for (int c = 0; c < 2; ++c)
      aq[c] = cat8(*reinterpret_cast<const v8h*>(ar + c * 32 + half * 16),
                   *reinterpret_cast<const v8h*>(ar + c * 32 + half * 16 + 8));
  }

  float mrun[8], lrun[8];
  v8f acc[4];
  for (int r = 0; r < 8; ++r) { mrun[r] = -__builtin_inff(); lrun[r] = 0.0f; }
  for (int t = 0; t < 4; ++t) acc[t] = zero8();

  for (int j0 = 0; j0 < m0 + 16; j0 += 32) {       // causal: keys <= m0+15
    v8f st[2];
    for (int t = 0; t < 2; ++t) {
      v8f s = zero8();
      const f16* br = kp + (size_t)(j0 + t * 16 + n) * HEADD;
      for (int c = 0; c < 2; ++c) {
        const v16h bk = *reinterpret_cast<const v16h*>(br + c * 32 + half * 16);
        s = wmma16(aq[c], bk, s);
      }
      st[t] = s;
    }
    for (int t = 0; t < 2; ++t)
      for (int r = 0; r < 8; ++r) {
        const int qi = m0 + r + 8 * half;
        const int kj = j0 + t * 16 + n;
        const float v = st[t][r] * 0.125f;          // 1/sqrt(64)
        st[t][r] = (kj > qi) ? -__builtin_inff() : v;
      }
    float p0[8], p1[8];
    for (int r = 0; r < 8; ++r) {
      float rm = fmaxf(st[0][r], st[1][r]);
      for (int off = 1; off < 16; off <<= 1) rm = fmaxf(rm, __shfl_xor(rm, off, 32));
      const float mn = fmaxf(mrun[r], rm);
      const float scale = __expf(mrun[r] - mn);
      mrun[r] = mn;
      p0[r] = __expf(st[0][r] - mn);
      p1[r] = __expf(st[1][r] - mn);
      float rs = p0[r] + p1[r];
      for (int off = 1; off < 16; off <<= 1) rs += __shfl_xor(rs, off, 32);
      lrun[r] = lrun[r] * scale + rs;
      for (int t = 0; t < 4; ++t) acc[t][r] *= scale;
    }
    for (int r = 0; r < 8; ++r) {
      plds[wave][r + 8 * half][n]      = (f16)p0[r];
      plds[wave][r + 8 * half][16 + n] = (f16)p1[r];
    }
    v16h pa;
    {
      const f16* pr = &plds[wave][n][0];
      pa = cat8(*reinterpret_cast<const v8h*>(pr + half * 16),
                *reinterpret_cast<const v8h*>(pr + half * 16 + 8));
    }
    for (int t = 0; t < 4; ++t) {
      const v16h bv = *reinterpret_cast<const v16h*>(
          vp + (size_t)(t * 16 + n) * S_LEN + j0 + half * 16);
      acc[t] = wmma16(pa, bv, acc[t]);
    }
  }

  for (int t = 0; t < 4; ++t)
    for (int r = 0; r < 8; ++r) {
      const int qi = m0 + r + 8 * half;
      const float o = acc[t][r] / lrun[r];
      const size_t row = (size_t)b * S_LEN + qi;
      oh[row * (NHEADS * HEADD) + hd * HEADD + t * 16 + n] = (f16)o;
    }
}

// ---------------------------------------------------------------------------
// Output projection: [4096,1024](f16) x Wp^T(f16) + bp -> fp32 d_out.
// 32x64 wave tile, B fragments reused across both M sub-tiles.
// ---------------------------------------------------------------------------
__global__ __launch_bounds__(256) void mha_out_gemm(const f16* __restrict__ oh,
                                                    const f16* __restrict__ WpT,
                                                    const float* __restrict__ bp,
                                                    float* __restrict__ out) {
  const int wid  = blockIdx.x * 8 + (threadIdx.x >> 5);   // 2048 waves
  const int lane = threadIdx.x & 31;
  const int n    = lane & 15;
  const int half = lane >> 4;
  const int ntile = wid & 15;
  const int mtile = wid >> 4;          // 0..127
  const int m0 = mtile * 32;
  const int n0 = ntile * 64;

  v8f acc[2][4];
  for (int mm = 0; mm < 2; ++mm)
    for (int t = 0; t < 4; ++t) acc[mm][t] = zero8();

  const f16* ar0 = oh + (size_t)(m0 + n) * E_DIM;
  const f16* ar1 = oh + (size_t)(m0 + 16 + n) * E_DIM;
  for (int kk = 0; kk < E_DIM; kk += 32) {
    const v16h af0 = cat8(*reinterpret_cast<const v8h*>(ar0 + kk + half * 16),
                          *reinterpret_cast<const v8h*>(ar0 + kk + half * 16 + 8));
    const v16h af1 = cat8(*reinterpret_cast<const v8h*>(ar1 + kk + half * 16),
                          *reinterpret_cast<const v8h*>(ar1 + kk + half * 16 + 8));
    for (int t = 0; t < 4; ++t) {
      const v16h bf = *reinterpret_cast<const v16h*>(
          WpT + (size_t)(n0 + t * 16 + n) * E_DIM + kk + half * 16);
      acc[0][t] = wmma16(af0, bf, acc[0][t]);
      acc[1][t] = wmma16(af1, bf, acc[1][t]);
    }
  }

  for (int mm = 0; mm < 2; ++mm)
    for (int t = 0; t < 4; ++t) {
      const int col = n0 + t * 16 + n;
      const float bias = bp[col];
      for (int r = 0; r < 8; ++r) {
        const int row = m0 + mm * 16 + r + 8 * half;
        out[(size_t)row * E_DIM + col] = acc[mm][t][r] + bias;
      }
    }
}

// ---------------------------------------------------------------------------
extern "C" void kernel_launch(void* const* d_in, const int* in_sizes, int n_in,
                              void* d_out, int out_size, void* d_ws, size_t ws_size,
                              hipStream_t stream) {
  const float* x  = (const float*)d_in[0];
  const float* Wq = (const float*)d_in[1];
  const float* Wk = (const float*)d_in[2];
  const float* Wv = (const float*)d_in[3];
  const float* Wp = (const float*)d_in[4];
  const float* bp = (const float*)d_in[5];
  float* out = (float*)d_out;

  char* ws = (char*)d_ws;
  f16* xh  = (f16*)(ws);                       // 8 MB  [4096,1024]
  f16* WqT = (f16*)(ws + (8ull  << 20));       // 2 MB  [1024,1024] K-major
  f16* WkT = (f16*)(ws + (10ull << 20));       // 2 MB
  f16* WvT = (f16*)(ws + (12ull << 20));       // 2 MB
  f16* WpT = (f16*)(ws + (14ull << 20));       // 2 MB
  f16* qh  = (f16*)(ws + (16ull << 20));       // 8 MB  [B,H,S,D]
  f16* kh  = (f16*)(ws + (24ull << 20));       // 8 MB  [B,H,S,D]
  f16* vT  = (f16*)(ws + (32ull << 20));       // 8 MB  [B,H,D,S]
  f16* oh  = (f16*)(ws + (40ull << 20));       // 8 MB  [B*S, H*D]

  mha_cvt_f16<<<4096, 256, 0, stream>>>(x, xh);
  dim3 tb(32, 8), tg(32, 32);
  mha_transpose_f16<<<tg, tb, 0, stream>>>(Wq, WqT);
  mha_transpose_f16<<<tg, tb, 0, stream>>>(Wk, WkT);
  mha_transpose_f16<<<tg, tb, 0, stream>>>(Wv, WvT);
  mha_transpose_f16<<<tg, tb, 0, stream>>>(Wp, WpT);

  // QKV: 3 * 128 * 16 = 6144 wave-tiles (32x64) / 8 waves per block
  mha_qkv_gemm<<<768, 256, 0, stream>>>(xh, WqT, WkT, WvT, qh, kh, vT);

  // attention: 2*16 (b,h) * 16 query blocks of 128 rows
  mha_attention<<<512, 256, 0, stream>>>(qh, kh, vT, oh);

  // projection: 128 * 16 = 2048 wave-tiles (32x64) / 8 per block
  mha_out_gemm<<<256, 256, 0, stream>>>(oh, WpT, bp, out);
}
